// GAT_E_to_R_51634096833136
// MI455X (gfx1250) — matrix operand order repaired and checked by reading, not compiled, not run on hardware
//
#include <hip/hip_runtime.h>
#include <hip/hip_bf16.h>

#define NN 100000
#define EE 500000
#define EHD 256
#define RHD 128
#define RR  1000

typedef __attribute__((ext_vector_type(16))) __bf16 v16bf;
typedef __attribute__((ext_vector_type(8)))  __bf16 v8bf;
typedef __attribute__((ext_vector_type(8)))  float  v8f;
typedef __attribute__((ext_vector_type(4)))  float  v4f;

// ---- ordered-uint encoding of float for atomic max ----
__device__ __forceinline__ unsigned fenc(float f) {
  unsigned u = __float_as_uint(f);
  return (u & 0x80000000u) ? ~u : (u | 0x80000000u);
}
__device__ __forceinline__ float fdec(unsigned e) {
  return __uint_as_float((e & 0x80000000u) ? (e & 0x7FFFFFFFu) : ~e);
}

__device__ __forceinline__ void fatomic_add(float* p, float v) {
  __hip_atomic_fetch_add(p, v, __ATOMIC_RELAXED, __HIP_MEMORY_SCOPE_AGENT);
}

// ---------------- init: zero out, -inf maxes, zero sums ----------------
__global__ void k_init(float* out, unsigned* m1, unsigned* m2, float* s1, float* s2) {
  int i = blockIdx.x * blockDim.x + threadIdx.x;
  if (i < RR * RHD) out[i] = 0.f;
  if (i < RR) {
    m1[i] = 0x007FFFFFu;   // fenc(-inf)
    m2[i] = 0x007FFFFFu;
    s1[i] = 0.f;
    s2[i] = 0.f;
  }
}

// ------- transpose + f32->bf16 convert: src[K][Ncols] -> dst[Ncols][K] -------
__global__ void k_wcvt(const float* __restrict__ src, __bf16* __restrict__ dst,
                       int K, int Ncols) {
  int i = blockIdx.x * blockDim.x + threadIdx.x;
  if (i >= K * Ncols) return;
  int n = i % Ncols, k = i / Ncols;
  dst[n * K + k] = (__bf16)src[(size_t)k * Ncols + n];
}

// ---------------- GEMM1: x_r_h = x_e@w_h, x_r_t = x_e@w_t ----------------
// One wave per 16-row tile of nodes. A kept resident in WMMA A layout.
__global__ __launch_bounds__(128)
void k_gemm1(const float* __restrict__ xe,
             const __bf16* __restrict__ whT, const __bf16* __restrict__ wtT,
             float* __restrict__ xrh, float* __restrict__ xrt,
             __bf16* __restrict__ xrh_bf, __bf16* __restrict__ xrt_bf) {
  const int lane = threadIdx.x & 31;
  const int wave = threadIdx.x >> 5;
  const int tile = blockIdx.x * 4 + wave;
  if (tile >= NN / 16) return;
  const int m0  = tile * 16;
  const int row = m0 + (lane & 15);
  const int hi  = lane >> 4;          // 0: lanes 0-15, 1: lanes 16-31
  const int bn  = lane & 15;

  // A as loaded here (16-bit, 16x32): lane<16 -> K 0..15; lane>=16 -> K 16..31 (same order as B)
  v16bf A[8];
  const float* xr = xe + (size_t)row * EHD;
  #pragma unroll
  for (int ks = 0; ks < 8; ++ks) {
    const int o1 = ks * 32 + hi * 16;
    v8f f1 = *(const v8f*)(xr + o1);
    v8f f2 = *(const v8f*)(xr + o1 + 8);
    #pragma unroll
    for (int j = 0; j < 8; ++j) {
      A[ks][j]     = (__bf16)f1[j];
      A[ks][j + 8] = (__bf16)f2[j];
    }
  }

  #pragma unroll
  for (int nt = 0; nt < 8; ++nt) {
    v8f ah = {}, at = {};
    // B layout (32x16): lanes 0-15 hold K 0-15 of col N=lane; lanes 16-31 hold K 16-31
    const __bf16* bh = whT + (size_t)(nt * 16 + bn) * EHD + hi * 16;
    const __bf16* bt = wtT + (size_t)(nt * 16 + bn) * EHD + hi * 16;
    #pragma unroll
    for (int ks = 0; ks < 8; ++ks) {
      v16bf Bh = *(const v16bf*)(bh + ks * 32);
      v16bf Bt = *(const v16bf*)(bt + ks * 32);
      ah = __builtin_amdgcn_wmma_f32_16x16x32_bf16(false, A[ks], false, Bh, (short)0, ah, false, false);
      at = __builtin_amdgcn_wmma_f32_16x16x32_bf16(false, A[ks], false, Bt, (short)0, at, false, false);
    }
    // D layout: vgpr g, half hi -> M = hi*8+g ; N = lane&15
    #pragma unroll
    for (int g = 0; g < 8; ++g) {
      const size_t m = (size_t)(m0 + hi * 8 + g);
      const int    n = nt * 16 + bn;
      xrh[m * RHD + n] = ah[g];
      xrt[m * RHD + n] = at[g];
      xrh_bf[m * RHD + n] = (__bf16)ah[g];
      xrt_bf[m * RHD + n] = (__bf16)at[g];
    }
  }
}

// ------- GEMM2: msg = x_r@r_w + b + x_r  (+ fused per-node attention dots) -------
__global__ __launch_bounds__(128)
void k_gemm2(const float* __restrict__ xrh, const float* __restrict__ xrt,
             const __bf16* __restrict__ xrh_bf, const __bf16* __restrict__ xrt_bf,
             const __bf16* __restrict__ rhwT, const __bf16* __restrict__ rtwT,
             const float* __restrict__ rhb, const float* __restrict__ rtb,
             const float* __restrict__ ah1, const float* __restrict__ ah2,
             const float* __restrict__ at1, const float* __restrict__ at2,
             float* __restrict__ msgh, float* __restrict__ msgt,
             float* __restrict__ sh1, float* __restrict__ sh2,
             float* __restrict__ st1, float* __restrict__ st2) {
  const int lane = threadIdx.x & 31;
  const int wave = threadIdx.x >> 5;
  const int tile = blockIdx.x * 4 + wave;
  if (tile >= NN / 16) return;
  const int m0  = tile * 16;
  const int row = m0 + (lane & 15);
  const int hi  = lane >> 4;
  const int bn  = lane & 15;

  v16bf Ah[4], At[4];
  const __bf16* arh = xrh_bf + (size_t)row * RHD;
  const __bf16* art = xrt_bf + (size_t)row * RHD;
  #pragma unroll
  for (int ks = 0; ks < 4; ++ks) {
    const int o1 = ks * 32 + hi * 16;
    v8bf h1 = *(const v8bf*)(arh + o1);
    v8bf h2 = *(const v8bf*)(arh + o1 + 8);
    v8bf t1 = *(const v8bf*)(art + o1);
    v8bf t2 = *(const v8bf*)(art + o1 + 8);
    #pragma unroll
    for (int j = 0; j < 8; ++j) {
      Ah[ks][j] = h1[j]; Ah[ks][j + 8] = h2[j];
      At[ks][j] = t1[j]; At[ks][j + 8] = t2[j];
    }
  }

  float p1[8] = {}, p2[8] = {}, p3[8] = {}, p4[8] = {};

  #pragma unroll
  for (int nt = 0; nt < 8; ++nt) {
    v8f ch = {}, ct = {};
    const __bf16* bh = rhwT + (size_t)(nt * 16 + bn) * RHD + hi * 16;
    const __bf16* bt = rtwT + (size_t)(nt * 16 + bn) * RHD + hi * 16;
    #pragma unroll
    for (int ks = 0; ks < 4; ++ks) {
      v16bf Bh = *(const v16bf*)(bh + ks * 32);
      v16bf Bt = *(const v16bf*)(bt + ks * 32);
      ch = __builtin_amdgcn_wmma_f32_16x16x32_bf16(false, Ah[ks], false, Bh, (short)0, ch, false, false);
      ct = __builtin_amdgcn_wmma_f32_16x16x32_bf16(false, At[ks], false, Bt, (short)0, ct, false, false);
    }
    const int n = nt * 16 + bn;
    const float bhb = rhb[n], btb = rtb[n];
    const float w1 = ah1[n], w2 = ah2[n], w3 = at1[n], w4 = at2[n];
    #pragma unroll
    for (int g = 0; g < 8; ++g) {
      const size_t m = (size_t)(m0 + hi * 8 + g);
      const float xh = xrh[m * RHD + n];
      const float xt = xrt[m * RHD + n];
      msgh[m * RHD + n] = ch[g] + bhb + xh;
      msgt[m * RHD + n] = ct[g] + btb + xt;
      p1[g] += xh * w1;   // x_r_h . a_h1
      p2[g] += xt * w2;   // x_r_t . a_h2
      p3[g] += xh * w3;   // x_r_h . a_t1
      p4[g] += xt * w4;   // x_r_t . a_t2
    }
  }

  // butterfly-reduce over the 16-lane half (n dimension)
  #pragma unroll
  for (int g = 0; g < 8; ++g) {
    #pragma unroll
    for (int d = 1; d < 16; d <<= 1) {
      p1[g] += __shfl_xor(p1[g], d, 32);
      p2[g] += __shfl_xor(p2[g], d, 32);
      p3[g] += __shfl_xor(p3[g], d, 32);
      p4[g] += __shfl_xor(p4[g], d, 32);
    }
  }
  if (bn == 0) {
    #pragma unroll
    for (int g = 0; g < 8; ++g) {
      const int m = m0 + hi * 8 + g;
      sh1[m] = p1[g]; sh2[m] = p2[g]; st1[m] = p3[g]; st2[m] = p4[g];
    }
  }
}

// ---------------- edge pass 1: logits + leaky-relu + segment max ----------------
__global__ void k_edge1(const int* __restrict__ hidx, const int* __restrict__ tidx,
                        const int* __restrict__ rel,
                        const float* __restrict__ sh1, const float* __restrict__ sh2,
                        const float* __restrict__ st1, const float* __restrict__ st2,
                        float* __restrict__ e1o, float* __restrict__ e2o,
                        unsigned* __restrict__ m1, unsigned* __restrict__ m2) {
  int e = blockIdx.x * blockDim.x + threadIdx.x;
  if (e >= EE) return;
  const int h = hidx[e], t = tidx[e], r = rel[e];
  float e1 = sh1[h] + sh2[t];
  float e2 = st1[h] + st2[t];
  e1 = e1 > 0.f ? e1 : 0.01f * e1;
  e2 = e2 > 0.f ? e2 : 0.01f * e2;
  e1o[e] = e1; e2o[e] = e2;
  atomicMax(&m1[r], fenc(e1));
  atomicMax(&m2[r], fenc(e2));
}

// ---------------- edge pass 2: exp + segment sum ----------------
__global__ void k_edge2(const int* __restrict__ rel,
                        const float* __restrict__ e1i, const float* __restrict__ e2i,
                        const unsigned* __restrict__ m1, const unsigned* __restrict__ m2,
                        float* __restrict__ x1o, float* __restrict__ x2o,
                        float* __restrict__ s1, float* __restrict__ s2) {
  int e = blockIdx.x * blockDim.x + threadIdx.x;
  if (e >= EE) return;
  const int r = rel[e];
  const float x1 = __expf(e1i[e] - fdec(m1[r]));
  const float x2 = __expf(e2i[e] - fdec(m2[r]));
  x1o[e] = x1; x2o[e] = x2;
  fatomic_add(&s1[r], x1);
  fatomic_add(&s2[r], x2);
}

// ---------------- scatter: out[rel] += a1*msg_h[h] + a2*msg_t[t] ----------------
__global__ __launch_bounds__(256)
void k_scatter(const int* __restrict__ hidx, const int* __restrict__ tidx,
               const int* __restrict__ rel,
               const float* __restrict__ x1, const float* __restrict__ x2,
               const float* __restrict__ s1, const float* __restrict__ s2,
               const float* __restrict__ msgh, const float* __restrict__ msgt,
               float* __restrict__ out) {
  const int wave = threadIdx.x >> 5;
  const int lane = threadIdx.x & 31;
  const int e = blockIdx.x * 8 + wave;
  if (e >= EE) return;
  const int h = hidx[e], t = tidx[e], r = rel[e];
  const float a1 = x1[e] / s1[r];
  const float a2 = x2[e] / s2[r];
  const int c = lane * 4;
  v4f mh = *(const v4f*)(msgh + (size_t)h * RHD + c);
  v4f mt = *(const v4f*)(msgt + (size_t)t * RHD + c);
  float* o = out + (size_t)r * RHD + c;
  #pragma unroll
  for (int i = 0; i < 4; ++i) fatomic_add(o + i, a1 * mh[i] + a2 * mt[i]);
}

extern "C" void kernel_launch(void* const* d_in, const int* in_sizes, int n_in,
                              void* d_out, int out_size, void* d_ws, size_t ws_size,
                              hipStream_t stream) {
  const float* x_e   = (const float*)d_in[0];
  const int*   eidx  = (const int*)d_in[1];
  const int*   rel   = (const int*)d_in[2];
  // d_in[3] = rel_all (unused by reference)
  const float* w_h   = (const float*)d_in[4];
  const float* w_t   = (const float*)d_in[5];
  const float* a_h1  = (const float*)d_in[6];
  const float* a_h2  = (const float*)d_in[7];
  const float* a_t1  = (const float*)d_in[8];
  const float* a_t2  = (const float*)d_in[9];
  const float* r_h_w = (const float*)d_in[10];
  const float* r_h_b = (const float*)d_in[11];
  const float* r_t_w = (const float*)d_in[12];
  const float* r_t_b = (const float*)d_in[13];
  float* out = (float*)d_out;
  const int* hidx = eidx;
  const int* tidx = eidx + EE;

  char* p = (char*)d_ws;
  auto take = [&](size_t bytes) -> char* {
    char* r = p;
    p += ((bytes + 255) & ~(size_t)255);
    return r;
  };
  __bf16* whT    = (__bf16*)take((size_t)EHD * RHD * 2);
  __bf16* wtT    = (__bf16*)take((size_t)EHD * RHD * 2);
  __bf16* rhwT   = (__bf16*)take((size_t)RHD * RHD * 2);
  __bf16* rtwT   = (__bf16*)take((size_t)RHD * RHD * 2);
  float*  xrh    = (float*)take((size_t)NN * RHD * 4);
  float*  xrt    = (float*)take((size_t)NN * RHD * 4);
  __bf16* xrh_bf = (__bf16*)take((size_t)NN * RHD * 2);
  __bf16* xrt_bf = (__bf16*)take((size_t)NN * RHD * 2);
  float*  msgh   = (float*)take((size_t)NN * RHD * 4);
  float*  msgt   = (float*)take((size_t)NN * RHD * 4);
  float*  sh1    = (float*)take((size_t)NN * 4);
  float*  sh2    = (float*)take((size_t)NN * 4);
  float*  st1    = (float*)take((size_t)NN * 4);
  float*  st2    = (float*)take((size_t)NN * 4);
  float*  e1b    = (float*)take((size_t)EE * 4);
  float*  e2b    = (float*)take((size_t)EE * 4);
  float*  x1b    = (float*)take((size_t)EE * 4);
  float*  x2b    = (float*)take((size_t)EE * 4);
  unsigned* m1   = (unsigned*)take((size_t)RR * 4);
  unsigned* m2   = (unsigned*)take((size_t)RR * 4);
  float*  s1b    = (float*)take((size_t)RR * 4);
  float*  s2b    = (float*)take((size_t)RR * 4);

  // 1. init accumulators / output
  k_init<<<(RR * RHD + 255) / 256, 256, 0, stream>>>(out, m1, m2, s1b, s2b);

  // 2. weight convert/transpose to bf16 B layout
  k_wcvt<<<(EHD * RHD + 255) / 256, 256, 0, stream>>>(w_h, whT, EHD, RHD);
  k_wcvt<<<(EHD * RHD + 255) / 256, 256, 0, stream>>>(w_t, wtT, EHD, RHD);
  k_wcvt<<<(RHD * RHD + 255) / 256, 256, 0, stream>>>(r_h_w, rhwT, RHD, RHD);
  k_wcvt<<<(RHD * RHD + 255) / 256, 256, 0, stream>>>(r_t_w, rtwT, RHD, RHD);

  // 3. GEMM1: x_r_h / x_r_t
  const int tiles = NN / 16;                      // 6250
  k_gemm1<<<(tiles + 3) / 4, 128, 0, stream>>>(x_e, whT, wtT, xrh, xrt, xrh_bf, xrt_bf);

  // 4. GEMM2: msg_h / msg_t + fused attention dot-products
  k_gemm2<<<(tiles + 3) / 4, 128, 0, stream>>>(xrh, xrt, xrh_bf, xrt_bf, rhwT, rtwT,
                                               r_h_b, r_t_b, a_h1, a_h2, a_t1, a_t2,
                                               msgh, msgt, sh1, sh2, st1, st2);

  // 5-6. segment softmax over rel
  k_edge1<<<(EE + 255) / 256, 256, 0, stream>>>(hidx, tidx, rel, sh1, sh2, st1, st2,
                                                e1b, e2b, m1, m2);
  k_edge2<<<(EE + 255) / 256, 256, 0, stream>>>(rel, e1b, e2b, m1, m2, x1b, x2b, s1b, s2b);

  // 7. fused scatter (out_h + out_t share rel)
  k_scatter<<<(EE + 7) / 8, 256, 0, stream>>>(hidx, tidx, rel, x1b, x2b, s1b, s2b,
                                              msgh, msgt, out);
}
